// KappaModel_46110768890490
// MI455X (gfx1250) — compile-verified, matrix operand order checked by reading, not run on hardware
//
#include <hip/hip_runtime.h>
#include <hip/hip_bf16.h>

typedef __attribute__((ext_vector_type(16))) _Float16 v16h;
typedef __attribute__((ext_vector_type(8)))  float    v8f;

#define BATCH 8
#define SEQL  2048
#define MTOK  (BATCH * SEQL)   // 16384
#define DIN   512
#define DSTATE 32
#define DTRANK 8
#define XPROJ_N (DTRANK + 2*DSTATE) // 72
#define MT 4                   // M-tiles (16 rows each) per wave in the GEMM

// ---------------- branchless activation helpers ----------------
__device__ __forceinline__ float elu_f(float v) {
    // exp argument clamped to <=0 so it is always cheap; select is a v_cndmask
    float e = __expf(fminf(v, 0.f)) - 1.f;
    return v > 0.f ? v : e;
}
__device__ __forceinline__ float tanh_f(float v) {
    float c = fminf(fmaxf(v, -10.f), 10.f);
    float e = __expf(2.f * c);
    return (e - 1.f) / (e + 1.f);
}
__device__ __forceinline__ float silu_f(float v) { return v / (1.f + __expf(-v)); }

template <int A>
__device__ __forceinline__ float act_pre(float v) {
    if constexpr (A == 1) return tanh_f(v);
    else if constexpr (A == 2) return elu_f(v);
    else return v;
}

// ---------------- WMMA GEMM: C[M,N] = act(A[M,K]) @ W[N,K]^T (+bias)(+elu) ----------------
// Each wave32 computes a 64x16 strip: 4 M-tiles sharing one B (weight) fragment per K-step.
// 8 waves / block. K % 32 == 0, M % 64 == 0, N arbitrary (tiles padded, guarded).
template <int PRE, int POST>
__global__ void k_gemm_wmma(const float* __restrict__ A, const float* __restrict__ W,
                            const float* __restrict__ bias, float* __restrict__ C,
                            int M, int N, int K)
{
    const int lane = threadIdx.x & 31;
    const int wave = threadIdx.x >> 5;
    const int ntiles = (N + 15) >> 4;
    const int msuper = M >> 6;                 // 64-row super-tiles
    const int total  = msuper * ntiles;
    const int tile   = blockIdx.x * 8 + wave;
    if (tile >= total) return;
    const int tm = tile / ntiles;              // super-tile row
    const int tn = tile - tm * ntiles;
    const int lane_lo = lane & 15;
    const int lane_hi = lane >> 4;

    const int bcol = tn * 16 + lane_lo;        // W row (output column)
    const bool wvalid = (bcol < N);
    const float* __restrict__ Wp = W + (size_t)(wvalid ? bcol : 0) * K;

    const float* __restrict__ Ap[MT];
#pragma unroll
    for (int t = 0; t < MT; ++t)
        Ap[t] = A + (size_t)(tm * 64 + t * 16 + lane_lo) * K;

    v8f acc[MT];
#pragma unroll
    for (int t = 0; t < MT; ++t) acc[t] = (v8f){};

    for (int k0 = 0; k0 < K; k0 += 32) {
        const int kb = k0 + lane_hi * 8;       // per-half-lane K base

        // ---- B fragment (weights), shared by all 4 M-tiles ----
        v16h bf;
        {
            float4 b0 = *reinterpret_cast<const float4*>(Wp + kb);
            float4 b1 = *reinterpret_cast<const float4*>(Wp + kb + 4);
            float4 b2 = *reinterpret_cast<const float4*>(Wp + kb + 16);
            float4 b3 = *reinterpret_cast<const float4*>(Wp + kb + 20);
            float s = wvalid ? 1.f : 0.f;
            bf[0]=(_Float16)(b0.x*s); bf[1]=(_Float16)(b0.y*s); bf[2]=(_Float16)(b0.z*s); bf[3]=(_Float16)(b0.w*s);
            bf[4]=(_Float16)(b1.x*s); bf[5]=(_Float16)(b1.y*s); bf[6]=(_Float16)(b1.z*s); bf[7]=(_Float16)(b1.w*s);
            bf[8]=(_Float16)(b2.x*s); bf[9]=(_Float16)(b2.y*s); bf[10]=(_Float16)(b2.z*s); bf[11]=(_Float16)(b2.w*s);
            bf[12]=(_Float16)(b3.x*s); bf[13]=(_Float16)(b3.y*s); bf[14]=(_Float16)(b3.z*s); bf[15]=(_Float16)(b3.w*s);
        }

#pragma unroll
        for (int t = 0; t < MT; ++t) {
            float4 a0 = *reinterpret_cast<const float4*>(Ap[t] + kb);
            float4 a1 = *reinterpret_cast<const float4*>(Ap[t] + kb + 4);
            float4 a2 = *reinterpret_cast<const float4*>(Ap[t] + kb + 16);
            float4 a3 = *reinterpret_cast<const float4*>(Ap[t] + kb + 20);
            v16h af;
            af[0]=(_Float16)act_pre<PRE>(a0.x); af[1]=(_Float16)act_pre<PRE>(a0.y);
            af[2]=(_Float16)act_pre<PRE>(a0.z); af[3]=(_Float16)act_pre<PRE>(a0.w);
            af[4]=(_Float16)act_pre<PRE>(a1.x); af[5]=(_Float16)act_pre<PRE>(a1.y);
            af[6]=(_Float16)act_pre<PRE>(a1.z); af[7]=(_Float16)act_pre<PRE>(a1.w);
            af[8]=(_Float16)act_pre<PRE>(a2.x); af[9]=(_Float16)act_pre<PRE>(a2.y);
            af[10]=(_Float16)act_pre<PRE>(a2.z); af[11]=(_Float16)act_pre<PRE>(a2.w);
            af[12]=(_Float16)act_pre<PRE>(a3.x); af[13]=(_Float16)act_pre<PRE>(a3.y);
            af[14]=(_Float16)act_pre<PRE>(a3.z); af[15]=(_Float16)act_pre<PRE>(a3.w);
            acc[t] = __builtin_amdgcn_wmma_f32_16x16x32_f16(false, af, false, bf,
                                                            (short)0, acc[t], false, false);
        }
    }

    const int n = tn * 16 + lane_lo;
    if (n < N) {
        const float bv = bias ? bias[n] : 0.f;
#pragma unroll
        for (int t = 0; t < MT; ++t) {
#pragma unroll
            for (int r = 0; r < 8; ++r) {
                const int m = tm * 64 + t * 16 + r + 8 * lane_hi;
                float v = acc[t][r] + bv;
                if constexpr (POST == 2) v = elu_f(v);
                C[(size_t)m * N + n] = v;
            }
        }
    }
}

// ---------------- input projection: h = x @ w0^T + b0  (K=3, N=128) ----------------
__global__ void k_proj_in(const float* __restrict__ x, const float* __restrict__ w0,
                          const float* __restrict__ b0, float* __restrict__ h)
{
    int idx = blockIdx.x * blockDim.x + threadIdx.x;
    if (idx >= MTOK * 128) return;
    int m = idx >> 7, o = idx & 127;
    float s = b0[o];
#pragma unroll
    for (int j = 0; j < 3; ++j) s += x[m * 3 + j] * w0[o * 3 + j];
    h[idx] = s;
}

// ---------------- depthwise causal conv(8) + bias + SiLU ----------------
// xz layout: (B*L, 1024) = [xc | z]; writes xc_act (B*L, 512)
__global__ void k_conv_silu(const float* __restrict__ xz, const float* __restrict__ cw,
                            const float* __restrict__ cb, float* __restrict__ xc)
{
    int idx = blockIdx.x * blockDim.x + threadIdx.x;
    if (idx >= MTOK * DIN) return;
    int d = idx & (DIN - 1);
    int t = idx >> 9;                 // flat token
    int l = t & (SEQL - 1);
    int b = t >> 11;
    float s = cb[d];
#pragma unroll
    for (int j = 0; j < 8; ++j) {
        int ll = l - 7 + j;
        if (ll >= 0)
            s += cw[d * 8 + j] * xz[((size_t)(b * SEQL + ll)) * 1024 + d];
    }
    xc[idx] = silu_f(s);
}

// ---------------- fused selective scan ----------------
// one thread per (b, d) channel; 32 states in registers; B_t/C_t staged in LDS;
// dt-proj + softplus + residual + SiLU gating fused. grid = (BATCH*8), block = 64.
__global__ void k_scan(const float* __restrict__ dbl, const float* __restrict__ xc,
                       const float* __restrict__ xz, const float* __restrict__ dtw,
                       const float* __restrict__ dtb, const float* __restrict__ alog,
                       const float* __restrict__ Dp, float* __restrict__ out)
{
    __shared__ float sBC[64];
    const int b = blockIdx.x >> 3;
    const int d = (blockIdx.x & 7) * 64 + threadIdx.x;

    float Areg[DSTATE], h[DSTATE];
#pragma unroll
    for (int n = 0; n < DSTATE; ++n) {
        Areg[n] = -__expf(alog[d * DSTATE + n]);
        h[n] = 0.f;
    }
    float wdt[DTRANK];
#pragma unroll
    for (int r = 0; r < DTRANK; ++r) wdt[r] = dtw[d * DTRANK + r];
    const float dt_bias = dtb[d];
    const float Dd = Dp[d];

    for (int l = 0; l < SEQL; ++l) {
        const size_t row = (size_t)(b * SEQL + l);
        const float* __restrict__ drow = dbl + row * XPROJ_N;
        __syncthreads();                       // protect sBC from previous step
        sBC[threadIdx.x] = drow[DTRANK + threadIdx.x];   // 64 floats: B(32) then C(32)
        __syncthreads();

        float dt = dt_bias;
#pragma unroll
        for (int r = 0; r < DTRANK; ++r) dt += wdt[r] * drow[r];
        // branchless softplus: log1p(exp(-|x|)) + max(x,0)
        dt = __logf(1.f + __expf(-fabsf(dt))) + fmaxf(dt, 0.f);

        const float x  = xc[row * DIN + d];
        const float dx = dt * x;
        float y = 0.f;
#pragma unroll
        for (int n = 0; n < DSTATE; ++n) {
            const float dA = __expf(dt * Areg[n]);
            h[n] = dA * h[n] + dx * sBC[n];
            y += h[n] * sBC[32 + n];
        }
        const float z = xz[row * 1024 + DIN + d];
        out[row * DIN + d] = (y + x * Dd) * silu_f(z);
    }
}

// ---------------- final: concat(x, mamba, lin) @ w2^T + b2 ----------------
__global__ void k_final(const float* __restrict__ x, const float* __restrict__ mamba,
                        const float* __restrict__ u, const float* __restrict__ w2,
                        const float* __restrict__ b2, float* __restrict__ out)
{
    int idx = blockIdx.x * blockDim.x + threadIdx.x;
    if (idx >= MTOK * 3) return;
    int m = idx / 3, o = idx - m * 3;
    const float* wr = w2 + o * 195;
    float s = b2[o];
#pragma unroll
    for (int j = 0; j < 3; ++j)   s += x[m * 3 + j]       * wr[j];
    for (int j = 0; j < 128; ++j) s += mamba[m * 128 + j] * wr[3 + j];
    for (int j = 0; j < 64; ++j)  s += u[m * 64 + j]      * wr[131 + j];
    out[idx] = s;
}

// ---------------- host side ----------------
template <int PRE, int POST>
static inline void launch_gemm(const float* A, const float* W, const float* bias, float* C,
                               int M, int N, int K, hipStream_t s)
{
    int ntiles = (N + 15) / 16;
    int tiles  = (M / 64) * ntiles;
    int blocks = (tiles + 7) / 8;
    k_gemm_wmma<PRE, POST><<<blocks, 256, 0, s>>>(A, W, bias, C, M, N, K);
}

extern "C" void kernel_launch(void* const* d_in, const int* in_sizes, int n_in,
                              void* d_out, int out_size, void* d_ws, size_t ws_size,
                              hipStream_t stream)
{
    const float* x       = (const float*)d_in[0];
    const float* w0      = (const float*)d_in[1];
    const float* b0      = (const float*)d_in[2];
    const float* m_in_w  = (const float*)d_in[3];   // (3,1024,128)
    const float* m_cw    = (const float*)d_in[4];   // (3,512,1,8)
    const float* m_cb    = (const float*)d_in[5];   // (3,512)
    const float* m_xpw   = (const float*)d_in[6];   // (3,72,512)
    const float* m_dtw   = (const float*)d_in[7];   // (3,512,8)
    const float* m_dtb   = (const float*)d_in[8];   // (3,512)
    const float* m_alog  = (const float*)d_in[9];   // (3,512,32)
    const float* m_D     = (const float*)d_in[10];  // (3,512)
    const float* m_outw  = (const float*)d_in[11];  // (3,128,512)
    const float* m_linw  = (const float*)d_in[12];  // (3,128,128)
    const float* m_linb  = (const float*)d_in[13];  // (3,128)
    const float* w1      = (const float*)d_in[14];  // (64,128)
    const float* b1      = (const float*)d_in[15];
    const float* ls_w    = (const float*)d_in[16];  // (12,64,64)
    const float* ls_b    = (const float*)d_in[17];
    const float* w2      = (const float*)d_in[18];  // (3,195)
    const float* b2      = (const float*)d_in[19];
    float* out = (float*)d_out;

    // workspace carve-up (all fp32)
    char* ws = (char*)d_ws;
    float* buf_h   = (float*)ws;                ws += (size_t)MTOK * 128  * 4;  // 8 MB
    float* buf_xz  = (float*)ws;                ws += (size_t)MTOK * 1024 * 4;  // 64 MB
    float* buf_xc  = (float*)ws;                ws += (size_t)MTOK * DIN  * 4;  // 32 MB
    float* buf_dbl = (float*)ws;                ws += (size_t)MTOK * XPROJ_N * 4; // 4.5 MB
    float* buf_ys  = (float*)ws;                ws += (size_t)MTOK * DIN  * 4;  // 32 MB
    float* buf_tmp = (float*)ws;                ws += (size_t)MTOK * 128  * 4;  // 8 MB
    float* buf_uA  = (float*)ws;                ws += (size_t)MTOK * 64   * 4;  // 4 MB
    float* buf_uB  = (float*)ws;                                                // 4 MB

    // 1) input projection
    k_proj_in<<<(MTOK * 128 + 255) / 256, 256, 0, stream>>>(x, w0, b0, buf_h);

    // 2) three mamba blocks
    for (int i = 0; i < 3; ++i) {
        const float* in_w = m_in_w + (size_t)i * 1024 * 128;
        // xz = tanh(h) @ in_w^T              (WMMA, tanh fused into A load)
        launch_gemm<1, 0>(buf_h, in_w, nullptr, buf_xz, MTOK, 1024, 128, stream);
        // depthwise conv + SiLU
        k_conv_silu<<<(MTOK * DIN + 255) / 256, 256, 0, stream>>>(
            buf_xz, m_cw + (size_t)i * DIN * 8, m_cb + (size_t)i * DIN, buf_xc);
        // dbl = xc @ xproj^T (N=72 -> 5 guarded tiles)
        launch_gemm<0, 0>(buf_xc, m_xpw + (size_t)i * XPROJ_N * DIN, nullptr, buf_dbl,
                          MTOK, XPROJ_N, DIN, stream);
        // fused dt-proj + softplus + scan + residual + SiLU gating
        k_scan<<<BATCH * 8, 64, 0, stream>>>(
            buf_dbl, buf_xc, buf_xz,
            m_dtw + (size_t)i * DIN * DTRANK, m_dtb + (size_t)i * DIN,
            m_alog + (size_t)i * DIN * DSTATE, m_D + (size_t)i * DIN, buf_ys);
        // tmp = elu(ys @ out_w^T)            (elu fused into store)
        launch_gemm<0, 2>(buf_ys, m_outw + (size_t)i * 128 * DIN, nullptr, buf_tmp,
                          MTOK, 128, DIN, stream);
        // h = tmp @ lin_w^T + lin_b
        launch_gemm<0, 0>(buf_tmp, m_linw + (size_t)i * 128 * 128, m_linb + (size_t)i * 128,
                          buf_h, MTOK, 128, 128, stream);
    }
    // buf_h now holds "mamba"

    // 3) head: 128 -> 64, then 12x (elu -> 64x64 + b)
    launch_gemm<0, 0>(buf_h, w1, b1, buf_uA, MTOK, 64, 128, stream);
    float* uin = buf_uA; float* uout = buf_uB;
    for (int j = 0; j < 12; ++j) {
        launch_gemm<2, 0>(uin, ls_w + (size_t)j * 64 * 64, ls_b + (size_t)j * 64,
                          uout, MTOK, 64, 64, stream);
        float* t = uin; uin = uout; uout = t;
    }
    // after 12 swaps the result is back in buf_uA (== uin)

    // 4) concat + final projection
    k_final<<<(MTOK * 3 + 255) / 256, 256, 0, stream>>>(x, buf_h, uin, w2, b2, out);

    (void)in_sizes; (void)n_in; (void)out_size; (void)ws_size;
}
